// DecoderLSTM_82944408421014
// MI455X (gfx1250) — matrix operand order repaired and checked by reading, not compiled, not run on hardware
//
#include <hip/hip_runtime.h>
#include <math.h>
#include <stdint.h>

// ---------------------------------------------------------------------------
// Problem dims (fixed by the reference)
// ---------------------------------------------------------------------------
#define BB 64
#define SS 64
#define EE 1024
#define UU 1024
#define AA 1024
#define VV 8192
#define TT 30

typedef __attribute__((ext_vector_type(16))) __bf16 v16bf;
typedef __attribute__((ext_vector_type(8)))  __bf16 v8bf;
typedef __attribute__((ext_vector_type(8)))  float  v8f;

__device__ __forceinline__ float sigmoidf_(float x) { return 1.0f / (1.0f + __expf(-x)); }
__device__ __forceinline__ __bf16 to_bf16(float f) { return (__bf16)f; }

// ---------------------------------------------------------------------------
// Pack a fp32 weight [Ksrc x Nsrc] (row-major, K outer) into bf16 WMMA B-tile
// layout.  One tile = 32(K) x 16(N) = 512 bf16 = 1024 B, stored lane-major
// (32 lanes x 16 elems).  Per ISA 7.12.2 (16-bit B matrix):
//   lanes 0-15  : K = ktile*32 + {0..7,16..23}   (elem e -> K offset e<8 ? e : e+8)
//   lanes 16-31 : K = ktile*32 + {8..15,24..31}
//   N = ntile*16 + (lane & 15)
// Tiles are stored **nt-major** over the FULL packed matrix [Ktot x Ntot]:
//   tile(nt, kt) at ((nt * Ktiles) + kt) * 512
// so one wave's B stream for a given N tile is fully contiguous in memory.
// (k_off, n_off) place this source block inside it (multiples of 32 / 16).
// ---------------------------------------------------------------------------
__global__ void pack_b_kernel(const float* __restrict__ src, __bf16* __restrict__ dst,
                              int Ksrc, int Nsrc, int k_off, int n_off, int ktiles_tot)
{
    size_t id = (size_t)blockIdx.x * blockDim.x + threadIdx.x;
    size_t total = (size_t)Ksrc * Nsrc;
    if (id >= total) return;
    int tiles_n = Nsrc >> 4;
    size_t tile = id >> 9;            // 512 elems per tile
    int r    = (int)(id & 511);
    int lane = r >> 4;
    int e    = r & 15;
    int kt_l = (int)(tile / tiles_n);
    int nt_l = (int)(tile % tiles_n);
    int k = kt_l * 32 + ((lane >= 16) ? 8 : 0) + ((e < 8) ? e : (e + 8));
    int n = nt_l * 16 + (lane & 15);
    float v = src[(size_t)k * Nsrc + n];
    size_t kt = (size_t)(k_off >> 5) + kt_l;
    size_t nt = (size_t)(n_off >> 4) + nt_l;
    dst[(nt * (size_t)ktiles_tot + kt) * 512 + (size_t)lane * 16 + e] = to_bf16(v);
}

__global__ void f32_to_bf16_kernel(const float* __restrict__ src, __bf16* __restrict__ dst, size_t n)
{
    size_t id = (size_t)blockIdx.x * blockDim.x + threadIdx.x;
    if (id < n) dst[id] = to_bf16(src[id]);
}

// ---------------------------------------------------------------------------
// bf16 WMMA GEMM:  C[M,N] = A[M,K] * Bpacked[K,N]   (fp32 accum)
//  - block = 1 wave (32 threads).  Wave owns N tile blockIdx.x and the full
//    64-row M strip blockIdx.y: 4 accumulators, so each B tile is loaded
//    exactly once and feeds 4 WMMAs (no redundant weight fetch).
//  - A: bf16 row-major, leading dim lda, in the ISA 16-bit A layout:
//    lanes 0-15 hold K {0..7,16..23}, lanes 16-31 {8..15,24..31}
//    -> two contiguous 16-byte loads per lane per tile per k-step.
//  - B: nt-major packed tiles -> one contiguous 32-byte load per lane per
//    k-step; prefetch ~16 KB ahead on the stream (global_prefetch_b8).
//  - C store follows the 32-bit C/D layout (VGPR r -> row r (+8 upper lanes)).
// ---------------------------------------------------------------------------
__global__ void gemm_bf16_wmma(const __bf16* __restrict__ A, int lda,
                               const __bf16* __restrict__ Bp, int ktiles_tot,
                               float* __restrict__ C, int ldc, int K)
{
    const int lane  = threadIdx.x & 31;
    const int nt    = blockIdx.x;
    const int m0    = blockIdx.y * 64;
    const int kbase = (lane >> 4) * 8;           // A layout half-select

    const __bf16* Bt  = Bp + (size_t)nt * ktiles_tot * 512 + (size_t)lane * 16;
    const __bf16* Ar  = A + (size_t)(m0 + (lane & 15)) * lda + kbase;
    const size_t  mstep = (size_t)16 * lda;

    v8f acc0 = {}, acc1 = {}, acc2 = {}, acc3 = {};
    const int ktn = K >> 5;

    for (int kt = 0; kt < ktn; ++kt) {
        v16bf b = *(const v16bf*)(Bt + (size_t)kt * 512);
        __builtin_prefetch(Bt + (size_t)(kt + 16) * 512, 0, 0);

        union { v16bf v; v8bf h[2]; } a;
        const __bf16* ak = Ar + kt * 32;

        a.h[0] = *(const v8bf*)(ak);
        a.h[1] = *(const v8bf*)(ak + 16);
        acc0 = __builtin_amdgcn_wmma_f32_16x16x32_bf16(false, a.v, false, b, (short)0, acc0, false, false);

        ak += mstep;
        a.h[0] = *(const v8bf*)(ak);
        a.h[1] = *(const v8bf*)(ak + 16);
        acc1 = __builtin_amdgcn_wmma_f32_16x16x32_bf16(false, a.v, false, b, (short)0, acc1, false, false);

        ak += mstep;
        a.h[0] = *(const v8bf*)(ak);
        a.h[1] = *(const v8bf*)(ak + 16);
        acc2 = __builtin_amdgcn_wmma_f32_16x16x32_bf16(false, a.v, false, b, (short)0, acc2, false, false);

        ak += mstep;
        a.h[0] = *(const v8bf*)(ak);
        a.h[1] = *(const v8bf*)(ak + 16);
        acc3 = __builtin_amdgcn_wmma_f32_16x16x32_bf16(false, a.v, false, b, (short)0, acc3, false, false);
    }

    const int col   = nt * 16 + (lane & 15);
    const int rsub  = (lane >> 4) * 8;
    float* Cc = C + col;
#pragma unroll
    for (int r = 0; r < 8; ++r) Cc[(size_t)(m0 +      rsub + r) * ldc] = acc0[r];
#pragma unroll
    for (int r = 0; r < 8; ++r) Cc[(size_t)(m0 + 16 + rsub + r) * ldc] = acc1[r];
#pragma unroll
    for (int r = 0; r < 8; ++r) Cc[(size_t)(m0 + 32 + rsub + r) * ldc] = acc2[r];
#pragma unroll
    for (int r = 0; r < 8; ++r) Cc[(size_t)(m0 + 48 + rsub + r) * ldc] = acc3[r];
}

// ---------------------------------------------------------------------------
// Bahdanau attention, one block per batch row b.
// score[s] = sum_a tanh(enc_proj[b,s,a] + q[b,a]) * wa[a] ; softmax over s ;
// context[b,e] = sum_s alpha[s] * enc_hidden[b,s,e]  -> bf16 into hc[:,1024:2048]
// ---------------------------------------------------------------------------
__global__ void attention_kernel(const float* __restrict__ enc_proj,  // [B,S,A]
                                 const float* __restrict__ xpre,      // [B,5120], q = cols 0..1023
                                 const float* __restrict__ wa,        // [A]
                                 const float* __restrict__ enc_hidden,// [B,S,E]
                                 __bf16* __restrict__ hc)             // [B,2048]
{
    const int b   = blockIdx.x;
    const int tid = threadIdx.x;                 // 256 threads
    __shared__ float sc[SS];
    __shared__ float red[256];

    const float* q = xpre + (size_t)b * (AA + 4 * UU);

    // partial score: 4 threads per s, 256 a-elements each
    const int s    = tid >> 2;
    const int part = tid & 3;
    const float* ep = enc_proj + ((size_t)b * SS + s) * AA;
    float acc = 0.0f;
    const int a0 = part * 256;
    for (int a = a0; a < a0 + 256; ++a)
        acc += tanhf(ep[a] + q[a]) * wa[a];
    red[tid] = acc;
    __syncthreads();
    if (part == 0) sc[s] = red[tid] + red[tid + 1] + red[tid + 2] + red[tid + 3];
    __syncthreads();

    if (tid == 0) {
        float m = sc[0];
        for (int i = 1; i < SS; ++i) m = fmaxf(m, sc[i]);
        float sum = 0.0f;
        for (int i = 0; i < SS; ++i) { sc[i] = __expf(sc[i] - m); sum += sc[i]; }
        float inv = 1.0f / sum;
        for (int i = 0; i < SS; ++i) sc[i] *= inv;
    }
    __syncthreads();

    const float* eh = enc_hidden + (size_t)b * SS * EE;
    for (int e = tid; e < EE; e += 256) {
        float c = 0.0f;
#pragma unroll 8
        for (int si = 0; si < SS; ++si)
            c += sc[si] * eh[(size_t)si * EE + e];
        hc[(size_t)b * 2048 + EE + e] = to_bf16(c);
    }
}

// ---------------------------------------------------------------------------
// Gate nonlinearities + LSTM state update.
// gateC[b, {f,i,o,g}*1024+u] = hidden@W*h + context@W*c ;
// xpre [b, 1024 + {f,i,o,g}*1024 + u] = out@W*x
// ---------------------------------------------------------------------------
__global__ void gate_update_kernel(const float* __restrict__ gateC,   // [B,4096]
                                   const float* __restrict__ xpre,    // [B,5120]
                                   const float* __restrict__ bf_,
                                   const float* __restrict__ bi_,
                                   const float* __restrict__ bo_,
                                   const float* __restrict__ bg_,
                                   float* __restrict__ state,         // [B,1024]
                                   __bf16* __restrict__ hc)           // [B,2048]
{
    const int id = blockIdx.x * blockDim.x + threadIdx.x;
    if (id >= BB * UU) return;
    const int b = id >> 10;
    const int u = id & 1023;
    const float* gc = gateC + (size_t)b * (4 * UU);
    const float* xp = xpre  + (size_t)b * (AA + 4 * UU);
    float f = sigmoidf_(gc[u]            + xp[AA + u]            + bf_[u]);
    float i = sigmoidf_(gc[UU + u]       + xp[AA + UU + u]       + bi_[u]);
    float o = sigmoidf_(gc[2 * UU + u]   + xp[AA + 2 * UU + u]   + bo_[u]);
    float g = sigmoidf_(gc[3 * UU + u]   + xp[AA + 3 * UU + u]   + bg_[u]);
    float st = f * state[id] + i * g;
    state[id] = st;
    hc[(size_t)b * 2048 + u] = to_bf16(tanhf(st) * o);
}

// ---------------------------------------------------------------------------
// Row softmax over V=8192 with bias; writes fp32 to d_out[b,t,:] and bf16
// activations for the next step's GEMM A operand.  One block per row.
// ---------------------------------------------------------------------------
__global__ void softmax_out_kernel(const float* __restrict__ logits,  // [B,V]
                                   const float* __restrict__ bp,      // [V]
                                   float* __restrict__ out_f32,       // [B,T,V]
                                   __bf16* __restrict__ out_bf,       // [B,V]
                                   int t)
{
    const int b   = blockIdx.x;
    const int tid = threadIdx.x;                 // 256
    __shared__ float red[256];
    const float* lr = logits + (size_t)b * VV;

    float m = -INFINITY;
    for (int v = tid; v < VV; v += 256) m = fmaxf(m, lr[v] + bp[v]);
    red[tid] = m; __syncthreads();
    for (int s = 128; s > 0; s >>= 1) { if (tid < s) red[tid] = fmaxf(red[tid], red[tid + s]); __syncthreads(); }
    m = red[0]; __syncthreads();

    float sum = 0.0f;
    for (int v = tid; v < VV; v += 256) sum += __expf(lr[v] + bp[v] - m);
    red[tid] = sum; __syncthreads();
    for (int s = 128; s > 0; s >>= 1) { if (tid < s) red[tid] += red[tid + s]; __syncthreads(); }
    const float inv = 1.0f / red[0];

    float*  op = out_f32 + ((size_t)b * TT + t) * VV;
    __bf16* ob = out_bf  + (size_t)b * VV;
    for (int v = tid; v < VV; v += 256) {
        float p = __expf(lr[v] + bp[v] - m) * inv;
        op[v] = p;
        ob[v] = to_bf16(p);
    }
}

// ---------------------------------------------------------------------------
// Host orchestration
// ---------------------------------------------------------------------------
static inline size_t cdiv(size_t a, size_t b) { return (a + b - 1) / b; }

extern "C" void kernel_launch(void* const* d_in, const int* in_sizes, int n_in,
                              void* d_out, int out_size, void* d_ws, size_t ws_size,
                              hipStream_t stream)
{
    (void)in_sizes; (void)n_in; (void)out_size; (void)ws_size;

    const float* initial_y  = (const float*)d_in[0];   // [B,1,V]
    const float* enc_hidden = (const float*)d_in[1];   // [B,S,E]
    const float* Wq  = (const float*)d_in[2];          // [V,A]
    const float* Wv  = (const float*)d_in[3];          // [E,A]
    const float* wa  = (const float*)d_in[4];          // [A,1]
    const float* Wfx = (const float*)d_in[5];
    const float* Wfh = (const float*)d_in[6];
    const float* Wfc = (const float*)d_in[7];
    const float* bf_ = (const float*)d_in[8];
    const float* Wix = (const float*)d_in[9];
    const float* Wih = (const float*)d_in[10];
    const float* Wic = (const float*)d_in[11];
    const float* bi_ = (const float*)d_in[12];
    const float* Wox = (const float*)d_in[13];
    const float* Woh = (const float*)d_in[14];
    const float* Woc = (const float*)d_in[15];
    const float* bo_ = (const float*)d_in[16];
    const float* Wgx = (const float*)d_in[17];
    const float* Wgh = (const float*)d_in[18];
    /* Wgc = d_in[19] is intentionally unused (source bug replicated: g-gate uses Wfc) */
    const float* bg_ = (const float*)d_in[20];
    const float* Wp  = (const float*)d_in[21];         // [U,V]
    const float* bp  = (const float*)d_in[22];         // [1,V]
    float* out = (float*)d_out;                        // [B,T,V]

    // ---- workspace carve-up --------------------------------------------
    uint8_t* ws = (uint8_t*)d_ws;
    size_t off = 0;
    auto alloc = [&](size_t bytes) -> uint8_t* {
        uint8_t* p = ws + off;
        off += (bytes + 255) & ~(size_t)255;
        return p;
    };
    __bf16* Bp_xcat = (__bf16*)alloc((size_t)VV * 5120 * 2);   // [Wq|Wfx|Wix|Wox|Wgx]  K=8192 N=5120
    __bf16* Bp_gate = (__bf16*)alloc((size_t)2048 * 4096 * 2); // [W*h;W*c] stacked     K=2048 N=4096
    __bf16* Bp_wp   = (__bf16*)alloc((size_t)UU * VV * 2);     // Wp                    K=1024 N=8192
    __bf16* Bp_wv   = (__bf16*)alloc((size_t)EE * AA * 2);     // Wv                    K=1024 N=1024
    __bf16* encB    = (__bf16*)alloc((size_t)BB * SS * EE * 2);
    float*  enc_proj= (float*) alloc((size_t)BB * SS * AA * 4);
    float*  xpre    = (float*) alloc((size_t)BB * 5120 * 4);
    float*  gateC   = (float*) alloc((size_t)BB * 4096 * 4);
    float*  logits  = (float*) alloc((size_t)BB * VV * 4);
    __bf16* out_bf  = (__bf16*)alloc((size_t)BB * VV * 2);
    __bf16* hc      = (__bf16*)alloc((size_t)BB * 2048 * 2);   // [hidden | context] bf16
    float*  state   = (float*) alloc((size_t)BB * UU * 4);

    const int PT = 256; // pack/convert block size
    auto pack = [&](const float* src, __bf16* dst, int K, int N, int ko, int no, int ktt) {
        pack_b_kernel<<<(unsigned)cdiv((size_t)K * N, PT), PT, 0, stream>>>(src, dst, K, N, ko, no, ktt);
    };

    // ---- Phase 0: weight packing (fp32 -> bf16 WMMA tiles, nt-major) ----
    pack(Wq,  Bp_xcat, VV, AA, 0, 0,         VV / 32);
    pack(Wfx, Bp_xcat, VV, UU, 0, AA,        VV / 32);
    pack(Wix, Bp_xcat, VV, UU, 0, AA + UU,   VV / 32);
    pack(Wox, Bp_xcat, VV, UU, 0, AA + 2*UU, VV / 32);
    pack(Wgx, Bp_xcat, VV, UU, 0, AA + 3*UU, VV / 32);

    pack(Wfh, Bp_gate, UU, UU, 0,    0,    2048 / 32);
    pack(Wfc, Bp_gate, EE, UU, 1024, 0,    2048 / 32);
    pack(Wih, Bp_gate, UU, UU, 0,    1024, 2048 / 32);
    pack(Wic, Bp_gate, EE, UU, 1024, 1024, 2048 / 32);
    pack(Woh, Bp_gate, UU, UU, 0,    2048, 2048 / 32);
    pack(Woc, Bp_gate, EE, UU, 1024, 2048, 2048 / 32);
    pack(Wgh, Bp_gate, UU, UU, 0,    3072, 2048 / 32);
    pack(Wfc, Bp_gate, EE, UU, 1024, 3072, 2048 / 32);  // source bug: g-gate reuses Wfc

    pack(Wp, Bp_wp, UU, VV, 0, 0, UU / 32);
    pack(Wv, Bp_wv, EE, AA, 0, 0, EE / 32);

    f32_to_bf16_kernel<<<(unsigned)cdiv((size_t)BB * SS * EE, PT), PT, 0, stream>>>(
        enc_hidden, encB, (size_t)BB * SS * EE);
    f32_to_bf16_kernel<<<(unsigned)cdiv((size_t)BB * VV, PT), PT, 0, stream>>>(
        initial_y, out_bf, (size_t)BB * VV);

    hipMemsetAsync(hc,    0, (size_t)BB * 2048 * 2, stream);
    hipMemsetAsync(state, 0, (size_t)BB * UU * 4,   stream);

    // enc_proj = enc_hidden @ Wv  : [4096,1024] = [4096,1024] x [1024,1024]
    gemm_bf16_wmma<<<dim3(AA / 16, (BB * SS) / 64), 32, 0, stream>>>(
        encB, EE, Bp_wv, EE / 32, enc_proj, AA, EE);

    // ---- Phase 1: the 30-step recurrence -------------------------------
    for (int t = 0; t < TT; ++t) {
        // [q | x-gates] = out @ Bp_xcat : [64,5120]
        gemm_bf16_wmma<<<dim3(5120 / 16, 1), 32, 0, stream>>>(
            out_bf, VV, Bp_xcat, VV / 32, xpre, 5120, VV);

        attention_kernel<<<BB, 256, 0, stream>>>(enc_proj, xpre, wa, enc_hidden, hc);

        // [f|i|o|g] h/c contributions = [hidden|context] @ Bp_gate : [64,4096]
        gemm_bf16_wmma<<<dim3(4096 / 16, 1), 32, 0, stream>>>(
            hc, 2048, Bp_gate, 2048 / 32, gateC, 4096, 2048);

        gate_update_kernel<<<(unsigned)cdiv((size_t)BB * UU, PT), PT, 0, stream>>>(
            gateC, xpre, bf_, bi_, bo_, bg_, state, hc);

        // logits = hidden @ Wp : [64,8192]   (A = hc cols 0..1023, lda=2048)
        gemm_bf16_wmma<<<dim3(VV / 16, 1), 32, 0, stream>>>(
            hc, 2048, Bp_wp, UU / 32, logits, VV, UU);

        softmax_out_kernel<<<BB, 256, 0, stream>>>(logits, bp, out, out_bf, t);
    }
}
